// MiAttention_54941221651288
// MI455X (gfx1250) — compile-verified, matrix operand order checked by reading, not run on hardware
//
#include <hip/hip_runtime.h>
#include <cstdint>
#include <cstddef>

// ---------------------------------------------------------------------------
// MI455X (gfx1250) attention block: QKV proj + RoPE + causal FA + O proj.
// All GEMM-like work on v_wmma_f32_16x16x32_bf16 (wave32 WMMA).
// ---------------------------------------------------------------------------

typedef __attribute__((ext_vector_type(16))) __bf16 v16bf;
typedef __attribute__((ext_vector_type(8)))  __bf16 v8bf;
typedef __attribute__((ext_vector_type(8)))  float  v8f;

#define S_LEN   4096
#define D_MODEL 2048
#define N_H     16
#define N_KVH   4
#define HEAD_D  128
#define HALF_D  64

// ---------------------------------------------------------------------------
// fp32 -> bf16 cast (grid-stride)
// ---------------------------------------------------------------------------
__global__ void cvt_f32_bf16(const float* __restrict__ src,
                             __bf16* __restrict__ dst, size_t n) {
  size_t i = (size_t)blockIdx.x * blockDim.x + threadIdx.x;
  size_t stride = (size_t)gridDim.x * blockDim.x;
  for (; i < n; i += stride) dst[i] = (__bf16)src[i];
}

// Load a 16x32 bf16 WMMA fragment for one lane.
// Lane halves: lanes 0-15 hold K {0..7,16..23}, lanes 16-31 hold K {8..15,24..31}.
// p must point at row_base + kBase + (laneHalf*8); 16B aligned.
__device__ __forceinline__ v16bf load_frag(const __bf16* __restrict__ p) {
  v8bf lo = *(const v8bf*)p;
  v8bf hi = *(const v8bf*)(p + 16);
  v16bf f;
#pragma unroll
  for (int e = 0; e < 8; ++e) { f[e] = lo[e]; f[e + 8] = hi[e]; }
  return f;
}

// ---------------------------------------------------------------------------
// C[M,N] (f32) = A[M,K] (bf16, row-major) * W[N,K]^T (bf16, row-major rows = N)
// One wave computes a 64x32 C tile: 4x2 grid of 16x16 WMMA accumulators.
// Register budget: 64 acc + 16 B-frag + 8 A-frag + addressing ~= 100 VGPRs,
// so the k-loop stays spill-free.
// ---------------------------------------------------------------------------
__global__ __launch_bounds__(256, 1)
void gemm_bf16_nt(const __bf16* __restrict__ A,
                  const __bf16* __restrict__ W,
                  float* __restrict__ C,
                  int M, int N, int K) {
  const int wave = (int)((blockIdx.x * blockDim.x + threadIdx.x) >> 5);
  const int lane = threadIdx.x & 31;
  const int tilesN = N >> 5;                  // 32-wide tiles
  const int tilesM = M >> 6;                  // 64-tall tiles
  if (wave >= tilesM * tilesN) return;        // uniform per wave
  const int tm = (wave / tilesN) * 64;
  const int tn = (wave % tilesN) * 32;
  const int hl = lane >> 4;                   // lane half
  const int r  = lane & 15;

  const __bf16* arow = A + (size_t)(tm + r) * K + hl * 8;
  const __bf16* brow = W + (size_t)(tn + r) * K + hl * 8;
  const size_t rowK16 = (size_t)16 * K;

  v8f acc[4][2] = {};

  for (int k0 = 0; k0 < K; k0 += 32) {
    v16bf b0 = load_frag(brow + k0);
    v16bf b1 = load_frag(brow + rowK16 + k0);
#pragma unroll
    for (int i = 0; i < 4; ++i) {
      const __bf16* pa = arow + (size_t)i * rowK16 + k0;
      v16bf a = load_frag(pa);
      __builtin_prefetch(pa + 32, 0, 1);      // warm next K chunk
      acc[i][0] = __builtin_amdgcn_wmma_f32_16x16x32_bf16(
          false, a, false, b0, (short)0, acc[i][0], false, false);
      acc[i][1] = __builtin_amdgcn_wmma_f32_16x16x32_bf16(
          false, a, false, b1, (short)0, acc[i][1], false, false);
    }
  }

  // C/D layout: lane half h, elem e -> row = tileRow + h*8 + e, col = tileCol + r
#pragma unroll
  for (int i = 0; i < 4; ++i) {
#pragma unroll
    for (int j = 0; j < 2; ++j) {
      const int row = tm + i * 16 + hl * 8;
      const int col = tn + j * 16 + r;
#pragma unroll
      for (int e = 0; e < 8; ++e)
        C[(size_t)(row + e) * N + col] = acc[i][j][e];
    }
  }
}

// ---------------------------------------------------------------------------
// RoPE + repack:  src [S][nh*HD] f32  ->  dst [nh][S][HD] bf16
// thread handles rotation pair (i, i+64) of one (s,h)
// ---------------------------------------------------------------------------
__global__ void rope_cvt(const float* __restrict__ src,
                         __bf16* __restrict__ dst,
                         const int* __restrict__ pos, int nh) {
  size_t idx = (size_t)blockIdx.x * blockDim.x + threadIdx.x;
  const size_t total = (size_t)S_LEN * nh * HALF_D;
  if (idx >= total) return;
  const int i = (int)(idx % HALF_D);
  size_t t = idx / HALF_D;
  const int h = (int)(t % nh);
  const int s = (int)(t / nh);

  const float p = (float)pos[s];
  // inv_freq[i] = 10000^(-2i/128) = exp(-ln(10000) * 2i/128)
  const float invf = __expf(-9.210340371976184f * (float)(2 * i) * (1.0f / 128.0f));
  const float th = p * invf;
  const float cs = __cosf(th);
  const float sn = __sinf(th);

  const float* rowp = src + (size_t)s * nh * HEAD_D + (size_t)h * HEAD_D;
  const float x1 = rowp[i];
  const float x2 = rowp[i + HALF_D];
  __bf16* drow = dst + ((size_t)h * S_LEN + s) * HEAD_D;
  drow[i]          = (__bf16)(x1 * cs - x2 * sn);
  drow[i + HALF_D] = (__bf16)(x2 * cs + x1 * sn);
}

// V repack: src [S][KVH*HD] f32 -> dst [KVH][S][HD] bf16
__global__ void v_cvt(const float* __restrict__ src, __bf16* __restrict__ dst) {
  size_t idx = (size_t)blockIdx.x * blockDim.x + threadIdx.x;
  const size_t total = (size_t)S_LEN * N_KVH * HEAD_D;
  if (idx >= total) return;
  const int d = (int)(idx % HEAD_D);
  size_t t = idx / HEAD_D;
  const int h = (int)(t % N_KVH);
  const int s = (int)(t / N_KVH);
  dst[((size_t)h * S_LEN + s) * HEAD_D + d] =
      (__bf16)src[(size_t)s * N_KVH * HEAD_D + (size_t)h * HEAD_D + d];
}

// ---------------------------------------------------------------------------
// Flash attention, causal. One wave owns 16 query rows of one head.
// Trick: compute S^T = K * Q^T so each lane owns one query column; softmax
// reductions are per-lane + one shfl_xor(16); P^T converts in place into the
// B operand of O^T += V^T * P (no cross-lane data movement).
// Q [H][S][HD] bf16, K/V [KVH][S][HD] bf16, O -> [S][H*HD] bf16.
// ---------------------------------------------------------------------------
__global__ __launch_bounds__(256, 1)
void attn_fa(const __bf16* __restrict__ Qb,
             const __bf16* __restrict__ Kb,
             const __bf16* __restrict__ Vb,
             __bf16* __restrict__ Ob) {
  const int lane = threadIdx.x & 31;
  const int gw = (int)(blockIdx.x * (blockDim.x >> 5) + (threadIdx.x >> 5));
  const int h     = gw / (S_LEN / 16);
  const int qBase = (gw % (S_LEN / 16)) * 16;
  const int kvh   = h / (N_H / N_KVH);

  const __bf16* Qh = Qb + (size_t)h   * S_LEN * HEAD_D;
  const __bf16* Kh = Kb + (size_t)kvh * S_LEN * HEAD_D;
  const __bf16* Vh = Vb + (size_t)kvh * S_LEN * HEAD_D;

  const int hl = lane >> 4;
  const int r  = lane & 15;
  const int q  = qBase + r;                 // this lane's query row

  // Q^T B-fragments over the 128-deep head dim (4 chunks of 32)
  v16bf qfrag[4];
#pragma unroll
  for (int c = 0; c < 4; ++c)
    qfrag[c] = load_frag(Qh + (size_t)q * HEAD_D + c * 32 + hl * 8);

  v8f o[8] = {};                            // O^T: 8 d-tiles x 16 q
  float m = -3.0e38f, l = 0.0f;
  const float scale = 0.08838834764831845f; // 1/sqrt(128)

  for (int kb = 0; kb < qBase + 16; kb += 32) {
    // ---- S^T (32 keys x 16 queries): A = K rows, B = Q^T -----------------
    v8f st[2] = {};
#pragma unroll
    for (int t = 0; t < 2; ++t)
#pragma unroll
      for (int c = 0; c < 4; ++c) {
        v16bf kf = load_frag(Kh + (size_t)(kb + t * 16 + r) * HEAD_D + c * 32 + hl * 8);
        st[t] = __builtin_amdgcn_wmma_f32_16x16x32_bf16(
            false, kf, false, qfrag[c], (short)0, st[t], false, false);
      }

    // ---- scale + causal mask + online softmax (in place in st) -----------
    float rowmax = -3.0e38f;
#pragma unroll
    for (int t = 0; t < 2; ++t)
#pragma unroll
      for (int e = 0; e < 8; ++e) {
        const int krow = kb + t * 16 + hl * 8 + e;
        float s = st[t][e] * scale;
        if (krow > q) s = -3.0e38f;
        st[t][e] = s;
        rowmax = fmaxf(rowmax, s);
      }
    rowmax = fmaxf(rowmax, __shfl_xor(rowmax, 16, 32)); // join lane-half pair
    const float mnew  = fmaxf(m, rowmax);
    const float alpha = __expf(m - mnew);

    v16bf pfrag;
    float psum = 0.0f;
#pragma unroll
    for (int t = 0; t < 2; ++t)
#pragma unroll
      for (int e = 0; e < 8; ++e) {
        const float p = __expf(st[t][e] - mnew);
        psum += p;
        pfrag[t * 8 + e] = (__bf16)p;       // P^T tile == B-operand layout
      }
    psum += __shfl_xor(psum, 16, 32);
    l = l * alpha + psum;
    m = mnew;

#pragma unroll
    for (int d = 0; d < 8; ++d)
#pragma unroll
      for (int e = 0; e < 8; ++e) o[d][e] *= alpha;

    // ---- O^T += V^T * P --------------------------------------------------
#pragma unroll
    for (int d = 0; d < 8; ++d) {
      const int dcol = d * 16 + r;
      v16bf vf;
#pragma unroll
      for (int e = 0; e < 8; ++e) {
        vf[e]     = Vh[(size_t)(kb + hl * 8 + e) * HEAD_D + dcol];
        vf[e + 8] = Vh[(size_t)(kb + 16 + hl * 8 + e) * HEAD_D + dcol];
      }
      o[d] = __builtin_amdgcn_wmma_f32_16x16x32_bf16(
          false, vf, false, pfrag, (short)0, o[d], false, false);
    }
  }

  // ---- normalize + store: out row q, cols h*128 + d ----------------------
  const float linv = 1.0f / l;
#pragma unroll
  for (int d = 0; d < 8; ++d) {
    const int dbase = d * 16 + hl * 8;
#pragma unroll
    for (int e = 0; e < 8; ++e)
      Ob[(size_t)q * (N_H * HEAD_D) + h * HEAD_D + dbase + e] =
          (__bf16)(o[d][e] * linv);
  }
}

// ---------------------------------------------------------------------------
// Host-side launch
// ---------------------------------------------------------------------------
extern "C" void kernel_launch(void* const* d_in, const int* in_sizes, int n_in,
                              void* d_out, int out_size, void* d_ws, size_t ws_size,
                              hipStream_t stream) {
  const float* hs  = (const float*)d_in[0];
  const float* Wq  = (const float*)d_in[1];
  const float* Wk  = (const float*)d_in[2];
  const float* Wv  = (const float*)d_in[3];
  const float* Wo  = (const float*)d_in[4];
  const int*   pos = (const int*)  d_in[5];
  float* out = (float*)d_out;

  char* ws = (char*)d_ws;
  size_t off = 0;
  auto alloc = [&](size_t bytes) -> void* {
    void* p = ws + off;
    off = (off + bytes + 255) & ~(size_t)255;
    return p;
  };

  const size_t KD = (size_t)N_KVH * HEAD_D;                 // 512
  __bf16* Xb  = (__bf16*)alloc((size_t)S_LEN * D_MODEL * 2);
  __bf16* Wqb = (__bf16*)alloc((size_t)D_MODEL * D_MODEL * 2);
  __bf16* Wkb = (__bf16*)alloc(KD * D_MODEL * 2);
  __bf16* Wvb = (__bf16*)alloc(KD * D_MODEL * 2);
  __bf16* Wob = (__bf16*)alloc((size_t)D_MODEL * D_MODEL * 2);
  float*  Qf  = (float*) alloc((size_t)S_LEN * D_MODEL * 4);
  float*  Kf  = (float*) alloc((size_t)S_LEN * KD * 4);
  float*  Vf  = (float*) alloc((size_t)S_LEN * KD * 4);
  __bf16* Qbh = (__bf16*)alloc((size_t)N_H   * S_LEN * HEAD_D * 2);
  __bf16* Kbh = (__bf16*)alloc((size_t)N_KVH * S_LEN * HEAD_D * 2);
  __bf16* Vbh = (__bf16*)alloc((size_t)N_KVH * S_LEN * HEAD_D * 2);
  __bf16* Ab  = (__bf16*)alloc((size_t)S_LEN * D_MODEL * 2);
  (void)ws_size; (void)in_sizes; (void)n_in; (void)out_size;

  // 1) cast everything to bf16
  cvt_f32_bf16<<<4096, 256, 0, stream>>>(hs, Xb, (size_t)S_LEN * D_MODEL);
  cvt_f32_bf16<<<4096, 256, 0, stream>>>(Wq, Wqb, (size_t)D_MODEL * D_MODEL);
  cvt_f32_bf16<<<1024, 256, 0, stream>>>(Wk, Wkb, KD * D_MODEL);
  cvt_f32_bf16<<<1024, 256, 0, stream>>>(Wv, Wvb, KD * D_MODEL);
  cvt_f32_bf16<<<4096, 256, 0, stream>>>(Wo, Wob, (size_t)D_MODEL * D_MODEL);

  // 2) QKV projections (wave tiles of 64x32, 8 waves per 256-thread block)
  {
    int waves = (S_LEN / 64) * (D_MODEL / 32);              // 4096
    gemm_bf16_nt<<<waves / 8, 256, 0, stream>>>(Xb, Wqb, Qf, S_LEN, D_MODEL, D_MODEL);
  }
  {
    int waves = (S_LEN / 64) * ((int)KD / 32);              // 1024
    gemm_bf16_nt<<<waves / 8, 256, 0, stream>>>(Xb, Wkb, Kf, S_LEN, (int)KD, D_MODEL);
    gemm_bf16_nt<<<waves / 8, 256, 0, stream>>>(Xb, Wvb, Vf, S_LEN, (int)KD, D_MODEL);
  }

  // 3) RoPE + head-major repack to bf16
  {
    size_t n = (size_t)S_LEN * N_H * HALF_D;
    rope_cvt<<<(unsigned)((n + 255) / 256), 256, 0, stream>>>(Qf, Qbh, pos, N_H);
  }
  {
    size_t n = (size_t)S_LEN * N_KVH * HALF_D;
    rope_cvt<<<(unsigned)((n + 255) / 256), 256, 0, stream>>>(Kf, Kbh, pos, N_KVH);
  }
  {
    size_t n = (size_t)S_LEN * N_KVH * HEAD_D;
    v_cvt<<<(unsigned)((n + 255) / 256), 256, 0, stream>>>(Vf, Vbh);
  }

  // 4) causal flash attention (one wave per 16 query rows per head)
  {
    int waves = N_H * (S_LEN / 16);                         // 4096
    attn_fa<<<waves / 8, 256, 0, stream>>>(Qbh, Kbh, Vbh, Ab);
  }

  // 5) output projection -> fp32 result
  {
    int waves = (S_LEN / 64) * (D_MODEL / 32);              // 4096
    gemm_bf16_nt<<<waves / 8, 256, 0, stream>>>(Ab, Wob, out, S_LEN, D_MODEL, D_MODEL);
  }
}
